// SLiGRU_54159537602779
// MI455X (gfx1250) — matrix operand order repaired and checked by reading, not compiled, not run on hardware
//
#include <hip/hip_runtime.h>
#include <hip/hip_bf16.h>
#include <math.h>

// ---------------------------------------------------------------------------
// Stacked LiGRU for MI455X (gfx1250, wave32, WMMA).
// bf16 WMMA (v_wmma_f32_16x16x32_bf16) for all matmuls, fp32 accumulate.
// A and B operands pre-converted to zero-padded bf16; B tiles staged via
// async global->LDS (double-buffered on ASYNCcnt).
// ---------------------------------------------------------------------------

typedef __bf16 bf16_t;
typedef __attribute__((ext_vector_type(16))) __bf16 v16bf;
typedef __attribute__((ext_vector_type(8)))  __bf16 v8bf;
typedef __attribute__((ext_vector_type(8)))  float  v8f;
typedef __attribute__((ext_vector_type(4)))  int    v4i;
typedef __attribute__((address_space(1))) v4i gv4i;  // global 16B chunk
typedef __attribute__((address_space(3))) v4i lv4i;  // LDS 16B chunk

#define LIGRU_B 32
#define LIGRU_T 1000
#define LIGRU_F 80
#define LIGRU_FP 96                  // F padded to K-step (32)
#define LIGRU_H 1024
#define LIGRU_G 2048                 // 2*H
#define LIGRU_M (LIGRU_B * LIGRU_T)  // 32000 rows
#define EPSF 1e-5f

// CDNA5 async global->LDS path (present on this toolchain; signature
// (v4i as1*, v4i as3*, imm offset, imm cpol)).
#if defined(__has_builtin)
#if __has_builtin(__builtin_amdgcn_global_load_async_to_lds_b128) && \
    __has_builtin(__builtin_amdgcn_s_wait_asynccnt)
#define USE_ASYNC_LDS 1
#endif
#endif
#ifndef USE_ASYNC_LDS
#define USE_ASYNC_LDS 0
#endif

__device__ __forceinline__ v8f wmma_bf16(v16bf a, v16bf b, v8f c) {
  // (neg_a, A, neg_b, B, c_mod, C, reuse_a, reuse_b)
  return __builtin_amdgcn_wmma_f32_16x16x32_bf16(false, a, false, b,
                                                 (short)0, c, false, false);
}

// ---------------------------------------------------------------------------
// fp32 [N,K] -> bf16 [N,Kp] with zero padding. grid = (ceil(Kp/256), N)
// Used for weights (W/U once per layer) and activations (x / h1).
// ---------------------------------------------------------------------------
__global__ void k_cvt_pad_bf16(const float* __restrict__ src,
                               bf16_t* __restrict__ dst, int K, int Kp) {
  const int n = blockIdx.y;
  const int k = blockIdx.x * blockDim.x + threadIdx.x;
  if (k < Kp)
    dst[(size_t)n * Kp + k] =
        (k < K) ? (bf16_t)src[(size_t)n * K + k] : (bf16_t)0.0f;
}

// ---------------------------------------------------------------------------
// Stage one 64(N) x 32(K) bf16 B tile into LDS.
// Async path: one 16B global_load_async_to_lds_b128 per thread.
// ---------------------------------------------------------------------------
__device__ __forceinline__ void stage_b_tile(const bf16_t* __restrict__ Bw,
                                             bf16_t* lds, int nbase, int Kp,
                                             int kb, int tid) {
#if USE_ASYNC_LDS
  const int col = tid >> 2, seg = tid & 3;
  const bf16_t* gp = Bw + (size_t)(nbase + col) * Kp + kb + seg * 8;
  bf16_t* lp = lds + col * 32 + seg * 8;
  __builtin_amdgcn_global_load_async_to_lds_b128((gv4i*)gp, (lv4i*)lp, 0, 0);
#else
  for (int idx = tid; idx < 64 * 32; idx += 256) {
    const int col = idx >> 5, k = idx & 31;
    lds[idx] = Bw[(size_t)(nbase + col) * Kp + kb + k];
  }
#endif
}

// ---------------------------------------------------------------------------
// C[M,N] = Abf[M,Kp](bf16) x Bw[N,Kp](bf16)^T  via WMMA bf16 (both padded).
// Block: 256 thr = 8 waves; block tile 128(M) x 64(N); K-step 32.
// M % 128 == 0 and Kp % 32 == 0 by construction -> no bounds checks anywhere.
// Double-buffered async B staging: issue tile kt+1, s_wait_asynccnt(1) for kt.
// ---------------------------------------------------------------------------
__global__ __launch_bounds__(256) void k_wgemm(const bf16_t* __restrict__ Abf,
                                               const bf16_t* __restrict__ Bw,
                                               float* __restrict__ C,
                                               int N, int Kp) {
  __shared__ __align__(64) bf16_t Bs[2][64 * 32];  // 2 x 4 KB
  const int tid = threadIdx.x;
  const int wv = tid >> 5;
  const int l = tid & 31;
  const int hlf = l >> 4;   // wave half
  const int ln = l & 15;
  const int mbase = blockIdx.x * 128 + wv * 16;
  const int nbase = blockIdx.y * 64;

  v8f acc[4] = {};

  const int ksteps = Kp >> 5;
  stage_b_tile(Bw, Bs[0], nbase, Kp, 0, tid);

  const bf16_t* arow = Abf + (size_t)(mbase + ln) * Kp;

  for (int kt = 0; kt < ksteps; ++kt) {
    const int kb = kt << 5;
    if (kt + 1 < ksteps) {
      stage_b_tile(Bw, Bs[(kt + 1) & 1], nbase, Kp, kb + 32, tid);
#if USE_ASYNC_LDS
      __builtin_amdgcn_s_wait_asynccnt(1);  // tile kt landed; kt+1 in flight
#endif
    } else {
#if USE_ASYNC_LDS
      __builtin_amdgcn_s_wait_asynccnt(0);
#endif
    }
    __syncthreads();
    const bf16_t* Bcur = Bs[kt & 1];

    // A fragment: two unconditional 16B loads (contiguous k = 16*half + s)
    v16bf af;
    {
      const bf16_t* ap = arow + kb + hlf * 16;
      const v8bf lo = *(const v8bf*)ap;
      const v8bf hi = *(const v8bf*)(ap + 8);
#pragma unroll
      for (int e = 0; e < 8; ++e) { af[e] = lo[e]; af[e + 8] = hi[e]; }
    }
#pragma unroll
    for (int j = 0; j < 4; ++j) {
      const v16bf bfr = *(const v16bf*)(Bcur + (j * 16 + ln) * 32 + hlf * 16);
      acc[j] = wmma_bf16(af, bfr, acc[j]);
    }
    __syncthreads();
  }

#pragma unroll
  for (int j = 0; j < 4; ++j) {
#pragma unroll
    for (int vi = 0; vi < 8; ++vi) {
      const int row = mbase + hlf * 8 + vi;
      const int col = nbase + j * 16 + ln;
      C[(size_t)row * N + col] = acc[j][vi];
    }
  }
}

// ---------------------------------------------------------------------------
// Batch-norm over rows: partial sums -> scale/shift -> fused apply
// ---------------------------------------------------------------------------
__global__ void k_zero2(float* a, float* b, int n) {
  int i = blockIdx.x * blockDim.x + threadIdx.x;
  if (i < n) { a[i] = 0.f; b[i] = 0.f; }
}

__global__ void k_bnstats(const float* __restrict__ w, float* sums,
                          float* sumsq, int Mrows) {
  const int col = blockIdx.x * blockDim.x + threadIdx.x;  // < 2048
  const int r0 = blockIdx.y * 128;
  const int r1 = (r0 + 128 < Mrows) ? r0 + 128 : Mrows;
  float s = 0.f, q = 0.f;
  for (int r = r0; r < r1; ++r) {
    const float v = w[(size_t)r * LIGRU_G + col];
    s += v; q += v * v;
  }
  atomicAdd(&sums[col], s);
  atomicAdd(&sumsq[col], q);
}

__global__ void k_bnfinal(float* sums, float* sumsq, const float* __restrict__ g,
                          const float* __restrict__ b, int Mrows) {
  const int col = blockIdx.x * blockDim.x + threadIdx.x;
  const float inv = 1.f / (float)Mrows;
  const float mu = sums[col] * inv;
  const float var = sumsq[col] * inv - mu * mu;
  const float scale = g[col] * rsqrtf(var + EPSF);
  sums[col] = scale;                 // reuse buffers: scale
  sumsq[col] = b[col] - mu * scale;  // shift
}

__global__ void k_bnapply(float* __restrict__ w, const float* __restrict__ scale,
                          const float* __restrict__ shift, size_t n) {
  size_t i = (size_t)blockIdx.x * blockDim.x + threadIdx.x;
  const size_t stride = (size_t)gridDim.x * blockDim.x;
  for (; i < n; i += stride) {
    const int col = (int)(i & (LIGRU_G - 1));
    w[i] = w[i] * scale[col] + shift[col];
  }
}

// ---------------------------------------------------------------------------
// Persistent recurrence: grid.x = 2 batch groups of 16 rows, 32 waves/WG.
// Per step: uh = h(16x1024) @ U^T(1024x2048) via WMMA (128 WMMA/wave),
// wave-shuffle LayerNorm, gates, h update. h kept in LDS (fp32 + bf16).
// LDS: uh 128KB + h 64KB + h_bf16 32KB + stats = ~224KB (< 320KB/WGP).
// ---------------------------------------------------------------------------
__global__ __launch_bounds__(1024) void k_ligru_rec(
    const float* __restrict__ wbuf,  // [B,T,G] BN'd gate inputs
    const bf16_t* __restrict__ Ubf,  // [G,H] bf16 (L2-resident)
    float* __restrict__ out,         // [B,T,H]
    int T) {
  extern __shared__ __align__(128) char smem_raw[];
  float* uh = (float*)smem_raw;            // 16*2048
  float* hf = uh + 16 * 2048;              // 16*1024 fp32 state
  bf16_t* hb = (bf16_t*)(hf + 16 * 1024);  // 16*1024 bf16 mirror (WMMA A)
  float* st = (float*)(hb + 16 * 1024);    // 16 x {mu, rstd}

  const int tid = threadIdx.x;
  const int wv = tid >> 5, l = tid & 31, hlf = l >> 4, ln = l & 15;
  const int g = blockIdx.x;  // batch group: rows b = g*16 + r

  for (int i = tid; i < 16 * 1024; i += 1024) {
    hf[i] = 0.f;
    hb[i] = (bf16_t)0.f;
  }
  __syncthreads();

  for (int t = 0; t < T; ++t) {
    // ---- phase 1: uh = h @ U^T (each wave owns 4 N-tiles of 16) ----
    v8f acc[4] = {};
#pragma unroll 1
    for (int kt = 0; kt < LIGRU_H / 32; ++kt) {
      const int kb = kt << 5;
      v16bf af;
      {
        const bf16_t* ap = hb + ln * LIGRU_H + kb + hlf * 16;
        const v8bf lo = *(const v8bf*)ap;
        const v8bf hi = *(const v8bf*)(ap + 8);
#pragma unroll
        for (int e = 0; e < 8; ++e) { af[e] = lo[e]; af[e + 8] = hi[e]; }
      }
#pragma unroll
      for (int j = 0; j < 4; ++j) {
        const int n = (wv * 4 + j) * 16 + ln;  // gate column
        // B fragment: 16 contiguous bf16 from U row n (32B aligned, L2 hit)
        const v16bf bfr =
            *(const v16bf*)(Ubf + (size_t)n * LIGRU_H + kb + hlf * 16);
        acc[j] = wmma_bf16(af, bfr, acc[j]);
      }
    }
#pragma unroll
    for (int j = 0; j < 4; ++j) {
      const int n = (wv * 4 + j) * 16 + ln;
#pragma unroll
      for (int vi = 0; vi < 8; ++vi)
        uh[(hlf * 8 + vi) * LIGRU_G + n] = acc[j][vi];
    }
    __syncthreads();

    // ---- phase 2: per-row LayerNorm stats (waves 0..15, one row each) ----
    if (wv < 16) {
      float s = 0.f, q = 0.f;
      const float* ur = uh + wv * LIGRU_G;
      for (int c = l; c < LIGRU_G; c += 32) {
        const float v = ur[c];
        s += v; q += v * v;
      }
#pragma unroll
      for (int m = 16; m >= 1; m >>= 1) {
        s += __shfl_xor(s, m, 32);
        q += __shfl_xor(q, m, 32);
      }
      if (l == 0) {
        const float mu = s * (1.f / (float)LIGRU_G);
        const float var = q * (1.f / (float)LIGRU_G) - mu * mu;
        st[wv * 2] = mu;
        st[wv * 2 + 1] = rsqrtf(var + EPSF);
      }
    }
    __syncthreads();

    // ---- phase 3: gates + state update + output store ----
#pragma unroll 1
    for (int it = 0; it < 16; ++it) {
      const int idx = tid + (it << 10);
      const int r = idx >> 10;       // batch row in group
      const int c = idx & (LIGRU_H - 1);
      const size_t wrow = ((size_t)(g * 16 + r) * T + t) * LIGRU_G;
      const float mu = st[r * 2], rstd = st[r * 2 + 1];
      const float a =
          wbuf[wrow + c] + (uh[r * LIGRU_G + c] - mu) * rstd;
      const float zi = wbuf[wrow + LIGRU_H + c] +
                       (uh[r * LIGRU_G + LIGRU_H + c] - mu) * rstd;
      const float z = 1.f / (1.f + __expf(-zi));
      const float hp = hf[idx];
      const float ra = a > 0.f ? a : 0.f;
      const float hn = z * hp + (1.f - z) * ra;
      hf[idx] = hn;
      hb[idx] = (bf16_t)hn;
      out[((size_t)(g * 16 + r) * T + t) * LIGRU_H + c] = hn;
      if (t + 1 < T)  // pull next step's gate inputs toward L2/WGP$
        __builtin_prefetch(&wbuf[wrow + LIGRU_G + c], 0, 0);
    }
    __syncthreads();
  }
}

// ---------------------------------------------------------------------------
extern "C" void kernel_launch(void* const* d_in, const int* in_sizes, int n_in,
                              void* d_out, int out_size, void* d_ws,
                              size_t ws_size, hipStream_t stream) {
  (void)in_sizes; (void)n_in; (void)out_size; (void)ws_size;
  const float* x = (const float*)d_in[0];
  const float* W1 = (const float*)d_in[1];
  const float* U1 = (const float*)d_in[2];
  const float* g1 = (const float*)d_in[3];
  const float* b1 = (const float*)d_in[4];
  const float* W2 = (const float*)d_in[5];
  const float* U2 = (const float*)d_in[6];
  const float* g2 = (const float*)d_in[7];
  const float* b2 = (const float*)d_in[8];
  float* out = (float*)d_out;

  // Workspace layout (~336 MB)
  char* ws = (char*)d_ws;
  const size_t WBUF_BYTES = (size_t)LIGRU_M * LIGRU_G * sizeof(float);
  const size_t ABF_BYTES = (size_t)LIGRU_M * LIGRU_H * sizeof(bf16_t);
  float* wbuf = (float*)ws;
  bf16_t* abf = (bf16_t*)(ws + WBUF_BYTES);            // activations, padded
  bf16_t* wbf = (bf16_t*)(ws + WBUF_BYTES + ABF_BYTES);
  bf16_t* ubf = wbf + (size_t)LIGRU_G * LIGRU_H;
  float* sums = (float*)(ubf + (size_t)LIGRU_G * LIGRU_H);
  float* sumsq = sums + LIGRU_G;

  const size_t SMEM_REC = (size_t)(16 * 2048 + 16 * 1024) * sizeof(float) +
                          (size_t)(16 * 1024) * sizeof(bf16_t) +
                          32 * sizeof(float);

  const dim3 gemm_grid(LIGRU_M / 128, LIGRU_G / 64);
  const dim3 stat_grid(LIGRU_G / 256, LIGRU_M / 128);
  const dim3 cvtW1_grid((LIGRU_FP + 255) / 256, LIGRU_G);   // W1: 2048 x 96
  const dim3 cvtWH_grid((LIGRU_H + 255) / 256, LIGRU_G);    // U/W2: 2048 x 1024
  const dim3 cvtA1_grid((LIGRU_FP + 255) / 256, LIGRU_M);   // x: 32000 x 96
  const dim3 cvtAH_grid((LIGRU_H + 255) / 256, LIGRU_M);    // h1: 32000 x 1024
  const size_t NW = (size_t)LIGRU_M * LIGRU_G;

  // ---------------- layer 1 ----------------
  k_cvt_pad_bf16<<<cvtW1_grid, 256, 0, stream>>>(W1, wbf, LIGRU_F, LIGRU_FP);
  k_cvt_pad_bf16<<<cvtWH_grid, 256, 0, stream>>>(U1, ubf, LIGRU_H, LIGRU_H);
  k_cvt_pad_bf16<<<cvtA1_grid, 256, 0, stream>>>(x, abf, LIGRU_F, LIGRU_FP);
  k_wgemm<<<gemm_grid, 256, 0, stream>>>(abf, wbf, wbuf, LIGRU_G, LIGRU_FP);
  k_zero2<<<LIGRU_G / 256, 256, 0, stream>>>(sums, sumsq, LIGRU_G);
  k_bnstats<<<stat_grid, 256, 0, stream>>>(wbuf, sums, sumsq, LIGRU_M);
  k_bnfinal<<<LIGRU_G / 256, 256, 0, stream>>>(sums, sumsq, g1, b1, LIGRU_M);
  k_bnapply<<<4096, 256, 0, stream>>>(wbuf, sums, sumsq, NW);
  k_ligru_rec<<<2, 1024, SMEM_REC, stream>>>(wbuf, ubf, out, LIGRU_T);

  // ---------------- layer 2 (h1 lives in d_out, consumed by cvt+GEMM) -----
  k_cvt_pad_bf16<<<cvtWH_grid, 256, 0, stream>>>(W2, wbf, LIGRU_H, LIGRU_H);
  k_cvt_pad_bf16<<<cvtWH_grid, 256, 0, stream>>>(U2, ubf, LIGRU_H, LIGRU_H);
  k_cvt_pad_bf16<<<cvtAH_grid, 256, 0, stream>>>(out, abf, LIGRU_H, LIGRU_H);
  k_wgemm<<<gemm_grid, 256, 0, stream>>>(abf, wbf, wbuf, LIGRU_G, LIGRU_H);
  k_zero2<<<LIGRU_G / 256, 256, 0, stream>>>(sums, sumsq, LIGRU_G);
  k_bnstats<<<stat_grid, 256, 0, stream>>>(wbuf, sums, sumsq, LIGRU_M);
  k_bnfinal<<<LIGRU_G / 256, 256, 0, stream>>>(sums, sumsq, g2, b2, LIGRU_M);
  k_bnapply<<<4096, 256, 0, stream>>>(wbuf, sums, sumsq, NW);
  k_ligru_rec<<<2, 1024, SMEM_REC, stream>>>(wbuf, ubf, out, LIGRU_T);
}
